// GVPCondenseMSA_4329327034445
// MI455X (gfx1250) — compile-verified, matrix operand order checked by reading, not run on hardware
//
#include <hip/hip_runtime.h>
#include <hip/hip_bf16.h>

typedef __attribute__((ext_vector_type(16))) _Float16 v16h;
typedef __attribute__((ext_vector_type(8)))  _Float16 v8h;
typedef __attribute__((ext_vector_type(8)))  float    v8f;

#define WMMA_F32_F16(a,b,c) \
  __builtin_amdgcn_wmma_f32_16x16x32_f16(false,(a),false,(b),(short)0,(c),false,false)

constexpr int B_   = 4;
constexpr int MAXS = 256;
constexpr int NT   = 128;   // terms
constexpr int TL   = 16;    // term length
constexpr int L_   = 2048;  // NT*TL
constexpr int NM   = 16;    // matches
constexpr int H_   = 128;
constexpr int NF_  = 9;
constexpr int KNB  = 8;
constexpr int NRBF = 16;
constexpr int CNODE = 192;  // 64 + 128
constexpr int MSGIN = 448;  // 2*192 + 64
constexpr int H2_  = 256;

// ---------------------------------------------------------------------------
// WMMA fragment loader.
// 16-bit A fragment (M x 32): lane = M (mod 16), halves 0..7 = K kb..kb+7,
// halves 8..15 = K kb+16..kb+23, kb = (lane>=16)?8:0.  B fragment (K x N) has
// the mirrored layout with lane = N, so the SAME loader works when the source
// is stored N-major [N][K].  Two 16-byte loads per fragment.
// ---------------------------------------------------------------------------
__device__ __forceinline__ v16h ld_frag(const _Float16* tile, int ld, int k0, int lane) {
  const _Float16* r = tile + (lane & 15) * ld + k0 + ((lane >> 4) << 3);
  v8h lo = *(const v8h*)r;
  v8h hi = *(const v8h*)(r + 16);
  v16h f = {};
#pragma unroll
  for (int i = 0; i < 8; ++i) { f[i] = lo[i]; f[i + 8] = hi[i]; }
  return f;
}

// ---------------------------------------------------------------------------
__global__ void zero_f32(float* p, long n) {
  long i = (long)blockIdx.x * blockDim.x + threadIdx.x;
  long stride = (long)gridDim.x * blockDim.x;
  for (; i < n; i += stride) p[i] = 0.f;
}

// W [K][N] f32 (row-major)  ->  WT [N][K] f16
__global__ void wt_kernel(const float* W, _Float16* WT, int K, int N) {
  int idx = blockIdx.x * blockDim.x + threadIdx.x;
  if (idx < K * N) {
    int k = idx / N, n = idx % N;
    WT[n * K + k] = (_Float16)W[k * N + n];
  }
}

// ---------------------------------------------------------------------------
// Condense: embed+concat+BN -> relu(x@W1+b1) -> tanh(@W2+b2) -> mean over NM
// One block per (b, pos): 16x128 @ 128x128 twice, 8 waves = 8 N-tiles.
// ---------------------------------------------------------------------------
__global__ __launch_bounds__(256) void condense_kernel(
    const int* X, const float* features, const unsigned char* mask,
    const float* embed, const float* g, const float* bta,
    const float* mean, const float* var,
    const _Float16* W1T, const float* b1,
    const _Float16* W2T, const float* b2,
    float* cond)
{
  int b = blockIdx.x >> 11, pos = blockIdx.x & 2047;
  int tid = threadIdx.x, lane = tid & 31, w = tid >> 5;
  __shared__ _Float16 sA[NM][H_];
  __shared__ _Float16 sH[NM][H_];

  for (int i = tid; i < NM * H_; i += 256) {
    int m = i >> 7, f = i & 127;
    float v;
    if (f < H_ - NF_) v = embed[X[(b * L_ + pos) * NM + m] * (H_ - NF_) + f];
    else              v = features[((b * L_ + pos) * NM + m) * NF_ + (f - (H_ - NF_))];
    v = (v - mean[f]) * rsqrtf(var[f] + 1e-5f) * g[f] + bta[f];
    sA[m][f] = (_Float16)v;
  }
  __syncthreads();

  v8f acc = {};
#pragma unroll
  for (int s = 0; s < 4; ++s) {
    v16h a  = ld_frag(&sA[0][0], H_, 32 * s, lane);
    v16h bf = ld_frag(W1T + w * 16 * H_, H_, 32 * s, lane);
    acc = WMMA_F32_F16(a, bf, acc);
  }
  int n = w * 16 + (lane & 15);
  float bias = b1[n];
#pragma unroll
  for (int r = 0; r < 8; ++r) {
    int M = r + ((lane >> 4) << 3);
    sH[M][n] = (_Float16)fmaxf(acc[r] + bias, 0.f);
  }
  __syncthreads();

  v8f acc2 = {};
#pragma unroll
  for (int s = 0; s < 4; ++s) {
    v16h a  = ld_frag(&sH[0][0], H_, 32 * s, lane);
    v16h bf = ld_frag(W2T + w * 16 * H_, H_, 32 * s, lane);
    acc2 = WMMA_F32_F16(a, bf, acc2);
  }
  float bias2 = b2[n];
  float part = 0.f;
#pragma unroll
  for (int r = 0; r < 8; ++r) part += tanhf(acc2[r] + bias2);
  part += __shfl_xor(part, 16, 32);   // combine the two M-halves
  if (lane < 16) {
    float nm = mask[b * L_ + pos] ? 0.f : 1.f;
    cond[((long)b * L_ + pos) * H_ + n] = part * (1.f / 16.f) * nm;
  }
}

// ---------------------------------------------------------------------------
// Geometry: per-term distances, top-8 neighbors, RBF edge features, Vgeo.
// ---------------------------------------------------------------------------
__global__ __launch_bounds__(128) void geom_kernel(
    const int* focuses, const float* coords,
    const float* Wv, const float* bv, const float* We, const float* be,
    float* Vgeo, float* Eb, int* relE, int* absE)
{
  int b = blockIdx.x >> 7, t = blockIdx.x & 127;
  int tid = threadIdx.x;
  __shared__ float sC12[TL][12];
  __shared__ float sD[TL][TL];
  __shared__ float sDk[TL][KNB];
  __shared__ int   sFoc[TL];

  if (tid < TL) sFoc[tid] = focuses[b * L_ + t * TL + tid];
  for (int i = tid; i < TL * 12; i += 128) {
    int r = i / 12, c = i % 12;
    int p = focuses[b * L_ + t * TL + r];
    sC12[r][c] = coords[((b * MAXS + p) * 4 + c / 3) * 3 + c % 3];
  }
  __syncthreads();

  for (int i = tid; i < TL * TL; i += 128) {
    int r = i >> 4, c = i & 15;
    float dx = sC12[r][3] - sC12[c][3];
    float dy = sC12[r][4] - sC12[c][4];
    float dz = sC12[r][5] - sC12[c][5];
    sD[r][c] = sqrtf(dx * dx + dy * dy + dz * dz + 1e-8f);
  }
  __syncthreads();

  if (tid < TL) {   // top-8 by selection: strict < keeps lowest index on ties
    bool used[TL];
    for (int j = 0; j < TL; ++j) used[j] = false;
    long base = ((long)(b * NT + t) * TL + tid) * KNB;
    for (int k = 0; k < KNB; ++k) {
      float best = 3.4e38f; int bi = 0;
      for (int j = 0; j < TL; ++j)
        if (!used[j] && sD[tid][j] < best) { best = sD[tid][j]; bi = j; }
      used[bi] = true;
      sDk[tid][k] = best;
      relE[base + k] = bi;
      absE[base + k] = sFoc[bi];
    }
  }
  __syncthreads();

  { // edge features: 128 threads == 128 (i,k) pairs
    int i = tid >> 3, k = tid & 7;
    float d = sDk[i][k];
    float rb[NRBF];
#pragma unroll
    for (int r = 0; r < NRBF; ++r) {
      float mu = 20.f * r / 15.f;
      float u = (d - mu) * (NRBF / 20.f);
      rb[r] = expf(-u * u);
    }
    long base = (((long)(b * NT + t) * TL + i) * KNB + k) * 64;
    for (int o = 0; o < 64; ++o) {
      float s = be[o];
#pragma unroll
      for (int r = 0; r < NRBF; ++r) s += rb[r] * We[r * 64 + o];
      Eb[base + o] = tanhf(s);
    }
  }
  for (int i = tid; i < TL * 64; i += 128) {
    int r = i >> 6, o = i & 63;
    float s = bv[o];
#pragma unroll
    for (int c = 0; c < 12; ++c) s += sC12[r][c] * Wv[c * 64 + o];
    Vgeo[((long)(b * NT + t) * TL + r) * 64 + o] = tanhf(s);
  }
}

// ---------------------------------------------------------------------------
// Message/node/edge: one block per term.  msg_in (128 rows x 448) built in
// LDS in 4 passes of 32 rows.  K loop outermost with 8 live accumulators
// (4 N-tiles x {Wmsg,Wedge}): each A fragment LDS read feeds 8 WMMAs.
// Neighbor mean folds in-register (each (i,n) owned by exactly one lane);
// node GEMM; float-atomic scatter into d_out accumulators.
// ---------------------------------------------------------------------------
__global__ __launch_bounds__(256) void msg_kernel(
    const float* cond, const float* Vgeo, const float* Eb,
    const int* relE, const int* absE, const int* focuses, const unsigned char* mask,
    const _Float16* WmT, const float* bm,
    const _Float16* WeT, const float* beE,
    const _Float16* WnT, const float* bnB,
    float* outAgg, float* outEdge, int* cntN, int* cntE)
{
  int b = blockIdx.x >> 7, t = blockIdx.x & 127;
  int tid = threadIdx.x, lane = tid & 31, w = tid >> 5;
  __shared__ _Float16 sIn[32][MSGIN];     // 28 KB (one 32-row pass)
  __shared__ _Float16 sNode[TL][MSGIN];   // 14 KB
  __shared__ float    sMean[TL][H2_];     // 16 KB
  __shared__ int   sFoc[TL], sRel[TL][KNB], sAbs[TL][KNB];
  __shared__ float sMask[TL];

  if (tid < TL) {
    sFoc[tid]  = focuses[b * L_ + t * TL + tid];
    sMask[tid] = mask[b * L_ + t * TL + tid] ? 0.f : 1.f;
  }
  if (tid < TL * KNB) {
    long base = (long)(b * NT + t) * TL * KNB + tid;
    sRel[tid >> 3][tid & 7] = relE[base];
    sAbs[tid >> 3][tid & 7] = absE[base];
  }
  for (int i = tid; i < TL * CNODE; i += 256) {  // batch_V rows
    int r = i / CNODE, c = i % CNODE;
    float v = (c < 64) ? Vgeo[((long)(b * NT + t) * TL + r) * 64 + c]
                       : cond[((long)b * L_ + t * TL + r) * H_ + (c - 64)];
    sNode[r][c] = (_Float16)v;
  }
  __syncthreads();

  int ml = w & 1, ng = w >> 1;   // wave -> local M-tile, N-tile group
  for (int pp = 0; pp < 4; ++pp) {
    for (int idx = tid; idx < 32 * MSGIN; idx += 256) {
      int lr = idx / MSGIN, c = idx % MSGIN;
      int grow = 32 * pp + lr;
      int i = grow >> 3, k = grow & 7;
      _Float16 v;
      if (c < CNODE)            v = sNode[i][c];
      else if (c < 2 * CNODE)   v = sNode[sRel[i][k]][c - CNODE];
      else v = (_Float16)Eb[(((long)(b * NT + t) * TL + i) * KNB + k) * 64 + (c - 2 * CNODE)];
      sIn[lr][c] = v;
    }
    __syncthreads();

    v8f aM[4], aE[4];
    const v8f vz = {};
#pragma unroll
    for (int q = 0; q < 4; ++q) { aM[q] = vz; aE[q] = vz; }

    for (int s = 0; s < 14; ++s) {
      v16h a = ld_frag(&sIn[ml * 16][0], MSGIN, 32 * s, lane);
#pragma unroll
      for (int q = 0; q < 4; ++q) {
        int ntile = ng * 4 + q;
        v16h bfm = ld_frag(WmT + ntile * 16 * MSGIN, MSGIN, 32 * s, lane);
        aM[q] = WMMA_F32_F16(a, bfm, aM[q]);
        v16h bfe = ld_frag(WeT + ntile * 16 * MSGIN, MSGIN, 32 * s, lane);
        aE[q] = WMMA_F32_F16(a, bfe, aE[q]);
      }
    }

#pragma unroll
    for (int q = 0; q < 4; ++q) {
      int ntile = ng * 4 + q;
      int n = ntile * 16 + (lane & 15);
      float biasM = bm[n], biasE = beE[n];
      float ssum = 0.f;
#pragma unroll
      for (int r = 0; r < 8; ++r) ssum += fmaxf(aM[q][r] + biasM, 0.f);
      sMean[4 * pp + 2 * ml + (lane >> 4)][n] = ssum * (1.f / KNB);
#pragma unroll
      for (int r = 0; r < 8; ++r) {
        int grow = 32 * pp + 16 * ml + r + ((lane >> 4) << 3);
        int i = grow >> 3, k = grow & 7;
        float e = fmaxf(aE[q][r] + biasE, 0.f) * sMask[i];
        long off = (((long)(b * MAXS + sFoc[i]) * MAXS) + sAbs[i][k]) * H2_ + n;
        atomicAdd(&outEdge[off], e);
      }
    }
    __syncthreads();
  }

  for (int idx = tid; idx < TL * H2_; idx += 256) {
    int i = idx >> 8, n = idx & 255;
    sNode[i][CNODE + n] = (_Float16)sMean[i][n];
  }
  __syncthreads();

  { // node GEMM: M=16, N=256, K=448; 2 N-tiles per wave, K outermost
    v8f accN[2];
    const v8f vz = {};
    accN[0] = vz; accN[1] = vz;
    for (int s = 0; s < 14; ++s) {
      v16h a = ld_frag(&sNode[0][0], MSGIN, 32 * s, lane);
#pragma unroll
      for (int q = 0; q < 2; ++q) {
        int ntile = w + 8 * q;
        v16h bf = ld_frag(WnT + ntile * 16 * MSGIN, MSGIN, 32 * s, lane);
        accN[q] = WMMA_F32_F16(a, bf, accN[q]);
      }
    }
#pragma unroll
    for (int q = 0; q < 2; ++q) {
      int ntile = w + 8 * q;
      int n = ntile * 16 + (lane & 15);
      float bias = bnB[n];
#pragma unroll
      for (int r = 0; r < 8; ++r) {
        int row = r + ((lane >> 4) << 3);
        float v = fmaxf(accN[q][r] + bias, 0.f) * sMask[row];
        atomicAdd(&outAgg[(long)(b * MAXS + sFoc[row]) * H2_ + n], v);
      }
    }
  }
  if (tid < TL) atomicAdd(&cntN[b * MAXS + sFoc[tid]], 1);
  if (tid < TL * KNB) {
    int i = tid >> 3, k = tid & 7;
    atomicAdd(&cntE[(b * MAXS + sFoc[i]) * MAXS + sAbs[i][k]], 1);
  }
}

// ---------------------------------------------------------------------------
__global__ void final_node_kernel(float* agg, const int* cntN, const int* seq_lens) {
  int idx = blockIdx.x * blockDim.x + threadIdx.x;
  if (idx >= B_ * MAXS * H2_) return;
  int b = idx / (MAXS * H2_);
  int p = (idx / H2_) % MAXS;
  int c = cntN[b * MAXS + p];
  if (p >= seq_lens[b]) c = 1;
  agg[idx] = agg[idx] / (float)c;
}

__global__ void final_edge_kernel(float* agge, const int* cntE) {
  long idx = (long)blockIdx.x * blockDim.x + threadIdx.x;
  if (idx >= (long)B_ * MAXS * MAXS * H2_) return;
  int c = cntE[idx >> 8];
  if (c < 1) c = 1;
  agge[idx] = agge[idx] / (float)c;
}

// ---------------------------------------------------------------------------
extern "C" void kernel_launch(void* const* d_in, const int* in_sizes, int n_in,
                              void* d_out, int out_size, void* d_ws, size_t ws_size,
                              hipStream_t stream) {
  const int*   X        = (const int*)d_in[0];
  const float* features = (const float*)d_in[1];
  const int*   seq_lens = (const int*)d_in[2];
  const int*   focuses  = (const int*)d_in[3];
  const unsigned char* src_mask = (const unsigned char*)d_in[5];
  const float* coords   = (const float*)d_in[8];
  const float* embed    = (const float*)d_in[10];
  const float* bn_g     = (const float*)d_in[11];
  const float* bn_b     = (const float*)d_in[12];
  const float* bn_m     = (const float*)d_in[13];
  const float* bn_v     = (const float*)d_in[14];
  const float* lin1_w   = (const float*)d_in[15];
  const float* lin1_b   = (const float*)d_in[16];
  const float* lin2_w   = (const float*)d_in[17];
  const float* lin2_b   = (const float*)d_in[18];
  const float* Wv       = (const float*)d_in[19];
  const float* bv       = (const float*)d_in[20];
  const float* We       = (const float*)d_in[21];
  const float* be       = (const float*)d_in[22];
  const float* Wmsg     = (const float*)d_in[23];
  const float* bmsg     = (const float*)d_in[24];
  const float* Wnode    = (const float*)d_in[25];
  const float* bnode    = (const float*)d_in[26];
  const float* Wedge    = (const float*)d_in[27];
  const float* bedge    = (const float*)d_in[28];

  char* ws = (char*)d_ws;
  size_t off = 0;
  auto alloc = [&](size_t bytes) -> void* {
    off = (off + 255) & ~(size_t)255;
    void* p = ws + off;
    off += bytes;
    return p;
  };
  float* cond  = (float*)alloc((size_t)B_ * L_ * H_ * 4);
  float* Vgeo  = (float*)alloc((size_t)B_ * NT * TL * 64 * 4);
  float* Eb    = (float*)alloc((size_t)B_ * NT * TL * KNB * 64 * 4);
  int*   relE  = (int*)alloc((size_t)B_ * NT * TL * KNB * 4);
  int*   absE  = (int*)alloc((size_t)B_ * NT * TL * KNB * 4);
  _Float16* W1T = (_Float16*)alloc((size_t)H_ * H_ * 2);
  _Float16* W2T = (_Float16*)alloc((size_t)H_ * H_ * 2);
  _Float16* WmT = (_Float16*)alloc((size_t)MSGIN * H2_ * 2);
  _Float16* WnT = (_Float16*)alloc((size_t)MSGIN * H2_ * 2);
  _Float16* WeT = (_Float16*)alloc((size_t)MSGIN * H2_ * 2);
  int* cntN = (int*)alloc((size_t)B_ * MAXS * 4);
  int* cntE = (int*)alloc((size_t)B_ * MAXS * MAXS * 4);

  float* outAgg  = (float*)d_out;
  float* outEdge = outAgg + (size_t)B_ * MAXS * H2_;
  long totalOut  = (long)B_ * MAXS * H2_ + (long)B_ * MAXS * MAXS * H2_;
  long totalEdge = (long)B_ * MAXS * MAXS * H2_;

  zero_f32<<<4096, 256, 0, stream>>>(outAgg, totalOut);
  zero_f32<<<64, 256, 0, stream>>>((float*)cntN, (long)B_ * MAXS);
  zero_f32<<<1024, 256, 0, stream>>>((float*)cntE, (long)B_ * MAXS * MAXS);

  wt_kernel<<<(H_ * H_ + 255) / 256, 256, 0, stream>>>(lin1_w, W1T, H_, H_);
  wt_kernel<<<(H_ * H_ + 255) / 256, 256, 0, stream>>>(lin2_w, W2T, H_, H_);
  wt_kernel<<<(MSGIN * H2_ + 255) / 256, 256, 0, stream>>>(Wmsg,  WmT, MSGIN, H2_);
  wt_kernel<<<(MSGIN * H2_ + 255) / 256, 256, 0, stream>>>(Wnode, WnT, MSGIN, H2_);
  wt_kernel<<<(MSGIN * H2_ + 255) / 256, 256, 0, stream>>>(Wedge, WeT, MSGIN, H2_);

  condense_kernel<<<B_ * L_, 256, 0, stream>>>(
      X, features, src_mask, embed, bn_g, bn_b, bn_m, bn_v,
      W1T, lin1_b, W2T, lin2_b, cond);

  geom_kernel<<<B_ * NT, 128, 0, stream>>>(
      focuses, coords, Wv, bv, We, be, Vgeo, Eb, relE, absE);

  msg_kernel<<<B_ * NT, 256, 0, stream>>>(
      cond, Vgeo, Eb, relE, absE, focuses, src_mask,
      WmT, bmsg, WeT, bedge, WnT, bnode,
      outAgg, outEdge, cntN, cntE);

  final_node_kernel<<<(B_ * MAXS * H2_ + 255) / 256, 256, 0, stream>>>(outAgg, cntN, seq_lens);
  final_edge_kernel<<<(int)((totalEdge + 255) / 256), 256, 0, stream>>>(outEdge, cntE);
}
